// UniCrystalFormer_25348896981385
// MI455X (gfx1250) — matrix operand order repaired and checked by reading, not compiled, not run on hardware
//
#include <hip/hip_runtime.h>
#include <hip/hip_bf16.h>
#include <math.h>

typedef __attribute__((ext_vector_type(16))) _Float16 v16h;
typedef __attribute__((ext_vector_type(8)))  _Float16 v8h;
typedef __attribute__((ext_vector_type(8)))  float    v8f;

static constexpr int   NN     = 8000;    // nodes
static constexpr int   NE     = 64000;   // edges
static constexpr int   NGR    = 64;      // graphs
static constexpr int   DD     = 128;     // hidden
static constexpr int   NH     = 4;       // heads
static constexpr int   CC     = 128;     // per-head
static constexpr int   HC     = 512;     // H*C
static constexpr int   C3     = 384;     // 3*C
static constexpr int   NLAY   = 3;
static constexpr int   NSTEP  = 3;
static constexpr int   CE     = 8000;    // matformer edge chunk
static constexpr int   NCHUNK = NE / CE;
static constexpr float FEPS   = 1e-5f;
static constexpr float FRAD   = 8.0f;

// fast reciprocal (v_rcp_f32) -- avoids IEEE divide expansion
__device__ __forceinline__ float frcp(float x) { return __builtin_amdgcn_rcpf(x); }
__device__ __forceinline__ float sigmf(float x) { return frcp(1.0f + __expf(-x)); }
__device__ __forceinline__ float siluf(float x) { return x * sigmf(x); }

// 128-thread block sum reduction (blockDim.x == 128)
__device__ __forceinline__ float bredsum(float v, float* red) {
  int t = threadIdx.x;
  red[t] = v; __syncthreads();
  for (int o = 64; o > 0; o >>= 1) {
    if (t < o) red[t] += red[t + o];
    __syncthreads();
  }
  float r = red[0]; __syncthreads();
  return r;
}

// async 16B global -> LDS copy (CDNA5 ASYNCcnt-tracked DMA, bypasses VGPRs)
__device__ __forceinline__ void async_ld_b128(void* lds, const void* gptr) {
  unsigned int  la = (unsigned int)(unsigned long long)(uintptr_t)lds;
  unsigned long long ga = (unsigned long long)(uintptr_t)gptr;
  asm volatile("global_load_async_to_lds_b128 %0, %1, off"
               :: "v"(la), "v"(ga) : "memory");
}
__device__ __forceinline__ void wait_asynccnt0() {
  asm volatile("s_wait_asynccnt 0x0" ::: "memory");
}

// ---------------------------------------------------------------------------
// WMMA GEMM:  Y[M,Nc] = act( A[M,K](lda) @ W[K,Nc] + bias ) * mul?
// Requirements (all call sites satisfy): M % 64 == 0, Nc % 64 == 0, K % 32 == 0,
// bias non-null, A/W/Y/mul 16-byte aligned. No bounds checks.
// block = 128 threads = 4 waves. Block tile 64x64, wave tile 32x32
// (2x2 v_wmma_f32_16x16x32_f16 per wave per K-step, f32 accumulate).
// A tile staged with GLOBAL_LOAD_ASYNC_TO_LDS_B128 (raw f32, no VGPR round
// trip); B tile staged transposed via registers (async DMA cannot transpose).
// f32 -> f16 conversion happens at fragment build (co-executes with WMMA).
// ACT: 0 none, 1 silu, 2 sigmoid. HASMUL: elementwise [M,Nc] multiplier.
// ---------------------------------------------------------------------------
template <int ACT, bool HASMUL>
__global__ __launch_bounds__(128) void k_gemm(
    const float* __restrict__ A, int lda,
    const float* __restrict__ W, const float* __restrict__ bias,
    float* __restrict__ Y, const float* __restrict__ mul,
    int M, int K, int Nc) {
  // f32 tiles, row stride 36 floats (144B = 9*16B: 16B-aligned b128 ops,
  // 36-bank stride -> conflict-free fragment reads across 16 lanes).
  __shared__ __align__(16) float As[64][36];    // A: [row][k]
  __shared__ __align__(16) float Bst[64][36];   // B transposed: [col][k]

  const int tid  = threadIdx.x;
  const int lane = tid & 31;
  const int wid  = tid >> 5;
  const int wm   = (wid >> 1) << 5;   // wave row offset in 64x64 block tile
  const int wn   = (wid & 1) << 5;    // wave col offset
  const int half = lane >> 4;
  const int mr   = lane & 15;
  const int rowBase = blockIdx.y << 6;
  const int colBase = blockIdx.x << 6;

  v8f acc00 = {0.f,0.f,0.f,0.f,0.f,0.f,0.f,0.f};
  v8f acc01 = acc00, acc10 = acc00, acc11 = acc00;

  for (int kb = 0; kb < K; kb += 32) {
    // ---- stage A: 64 rows x 32 K as raw f32 via async DMA to LDS ----
#pragma unroll
    for (int it = 0; it < 4; ++it) {
      int idx = it * 128 + tid;           // 0..511 16B chunks
      int r = idx >> 3, q = idx & 7;      // row 0..63, 4-float group 0..7
      async_ld_b128(&As[r][q * 4],
                    A + (size_t)(rowBase + r) * lda + kb + q * 4);
    }
    // ---- stage B transposed: 32 K x 64 cols (register path) ----
#pragma unroll
    for (int it = 0; it < 2; ++it) {
      int idx = it * 128 + tid;           // 0..255
      int r = idx >> 3, q = idx & 7;      // k-row 0..31, 8-col group 0..7
      const float4* pb =
          (const float4*)(W + (size_t)(kb + r) * Nc + colBase + q * 8);
      float4 u = pb[0], w4 = pb[1];
      Bst[q * 8 + 0][r] = u.x;
      Bst[q * 8 + 1][r] = u.y;
      Bst[q * 8 + 2][r] = u.z;
      Bst[q * 8 + 3][r] = u.w;
      Bst[q * 8 + 4][r] = w4.x;
      Bst[q * 8 + 5][r] = w4.y;
      Bst[q * 8 + 6][r] = w4.z;
      Bst[q * 8 + 7][r] = w4.w;
    }
    wait_asynccnt0();       // this wave's DMA writes to LDS are visible
    __syncthreads();        // all waves' stores published

    // ---- fragments (ISA 7.12.2 layouts; contiguous 16B chunks + cvt) ----
    v16h af[2], bf[2];
#pragma unroll
    for (int i = 0; i < 2; ++i) {
      const float* ar = &As[wm + i * 16 + mr][0];
      float4 c0 = *(const float4*)&ar[half * 8];        // K = half*8 .. +3
      float4 c1 = *(const float4*)&ar[half * 8 + 4];    // K = half*8+4 .. +7
      float4 c2 = *(const float4*)&ar[16 + half * 8];   // K = 16+half*8 ..
      float4 c3 = *(const float4*)&ar[16 + half * 8 + 4];
      v16h f;
      f[0]=(_Float16)c0.x; f[1]=(_Float16)c0.y; f[2]=(_Float16)c0.z; f[3]=(_Float16)c0.w;
      f[4]=(_Float16)c1.x; f[5]=(_Float16)c1.y; f[6]=(_Float16)c1.z; f[7]=(_Float16)c1.w;
      f[8]=(_Float16)c2.x; f[9]=(_Float16)c2.y; f[10]=(_Float16)c2.z; f[11]=(_Float16)c2.w;
      f[12]=(_Float16)c3.x; f[13]=(_Float16)c3.y; f[14]=(_Float16)c3.z; f[15]=(_Float16)c3.w;
      af[i] = f;
    }
#pragma unroll
    for (int j = 0; j < 2; ++j) {
      const float* br = &Bst[wn + j * 16 + mr][0];
      float4 c0 = *(const float4*)&br[half * 8];        // K = half*8 .. +3
      float4 c1 = *(const float4*)&br[half * 8 + 4];
      float4 c2 = *(const float4*)&br[16 + half * 8];
      float4 c3 = *(const float4*)&br[16 + half * 8 + 4];  // .. K = 16+half*8+7
      v16h f;
      f[0]=(_Float16)c0.x; f[1]=(_Float16)c0.y; f[2]=(_Float16)c0.z; f[3]=(_Float16)c0.w;
      f[4]=(_Float16)c1.x; f[5]=(_Float16)c1.y; f[6]=(_Float16)c1.z; f[7]=(_Float16)c1.w;
      f[8]=(_Float16)c2.x; f[9]=(_Float16)c2.y; f[10]=(_Float16)c2.z; f[11]=(_Float16)c2.w;
      f[12]=(_Float16)c3.x; f[13]=(_Float16)c3.y; f[14]=(_Float16)c3.z; f[15]=(_Float16)c3.w;
      bf[j] = f;
    }
    acc00 = __builtin_amdgcn_wmma_f32_16x16x32_f16(false, af[0], false, bf[0],
                                                   (short)0, acc00, false, false);
    acc01 = __builtin_amdgcn_wmma_f32_16x16x32_f16(false, af[0], false, bf[1],
                                                   (short)0, acc01, false, false);
    acc10 = __builtin_amdgcn_wmma_f32_16x16x32_f16(false, af[1], false, bf[0],
                                                   (short)0, acc10, false, false);
    acc11 = __builtin_amdgcn_wmma_f32_16x16x32_f16(false, af[1], false, bf[1],
                                                   (short)0, acc11, false, false);
    __syncthreads();
  }

  // ---- epilogue: C layout m = r + half*8 (per 16x16 tile), n = mr ----
  v8f* accs[4] = {&acc00, &acc01, &acc10, &acc11};
#pragma unroll
  for (int i = 0; i < 2; ++i) {
#pragma unroll
    for (int j = 0; j < 2; ++j) {
      v8f& a = *accs[i * 2 + j];
      int nn = colBase + wn + j * 16 + mr;
      float bv = bias[nn];
#pragma unroll
      for (int r = 0; r < 8; ++r) {
        int mm = rowBase + wm + i * 16 + (half << 3) + r;
        float v = a[r] + bv;
        if (ACT == 1) v = siluf(v);
        else if (ACT == 2) v = sigmf(v);
        if (HASMUL) v *= mul[(size_t)mm * Nc + nn];
        Y[(size_t)mm * Nc + nn] = v;
      }
    }
  }
}

// ---------------------------------------------------------------------------
// Elementwise / reduction kernels
// ---------------------------------------------------------------------------
__global__ void k_fill(float* p, float v, long long n) {
  long long i = (long long)blockIdx.x * blockDim.x + threadIdx.x;
  long long s = (long long)gridDim.x * blockDim.x;
  for (; i < n; i += s) p[i] = v;
}

__global__ void k_counts(const int* batch, float* counts) {
  int i = blockIdx.x * 256 + threadIdx.x;
  if (i < NN) atomicAdd(&counts[batch[i]], 1.0f);
}
__global__ void k_counts_clamp(float* counts) {
  int i = threadIdx.x;
  if (i < NGR) counts[i] = fmaxf(counts[i], 1.0f);
}

// atom encoder front: emb lookup + megnet linear (K=16, done in VALU)
__global__ void k_atom_pre(const int* an, const float* mge, const float* emb,
                           const float* Wm, const float* bm, float* cat) {
  int n = blockIdx.x, c = threadIdx.x;
  float ev = emb[(size_t)an[n] * DD + c];
  float acc = bm[c];
#pragma unroll
  for (int j = 0; j < 16; ++j) acc += mge[(size_t)n * 16 + j] * Wm[j * DD + c];
  cat[(size_t)n * 256 + c]       = ev;
  cat[(size_t)n * 256 + 128 + c] = acc;
}

__global__ void k_atom_fuse(const float* g, const float* cat, float* fused) {
  int n = blockIdx.x, c = threadIdx.x;
  float gv = g[(size_t)n * DD + c];
  fused[(size_t)n * DD + c] = gv * cat[(size_t)n * 256 + c] +
                              (1.f - gv) * cat[(size_t)n * 256 + 128 + c];
}

// generic row LayerNorm (blockDim=128, L multiple of 128). act: 0 none, 2 sigmoid
__global__ void k_ln_row(const float* X, float* Y, const float* g, const float* b,
                         int L, int act) {
  __shared__ float red[128];
  int row = blockIdx.x;
  const float* x = X + (size_t)row * L;
  float* y = Y + (size_t)row * L;
  float invL = frcp((float)L);
  float s = 0.f;
  for (int i = threadIdx.x; i < L; i += 128) s += x[i];
  float mean = bredsum(s, red) * invL;
  s = 0.f;
  for (int i = threadIdx.x; i < L; i += 128) { float d = x[i] - mean; s += d * d; }
  float rstd = rsqrtf(bredsum(s, red) * invL + FEPS);
  for (int i = threadIdx.x; i < L; i += 128) {
    float v = g[i] * (x[i] - mean) * rstd + b[i];
    if (act == 2) v = sigmf(v);
    y[i] = v;
  }
}

__global__ void k_rbf(const float* evec, float* rbff, float* env) {
  int e = blockIdx.x, c = threadIdx.x;
  float dx = evec[(size_t)e * 3], dy = evec[(size_t)e * 3 + 1], dz = evec[(size_t)e * 3 + 2];
  float dist = sqrtf(dx * dx + dy * dy + dz * dz);
  const float delta = FRAD / (float)(128 - 1);
  float dd = dist - (float)c * delta;
  rbff[(size_t)e * DD + c] = __expf(-(dd * dd) * (1.0f / (delta * delta)));
  if (c == 0) {
    float dcl = fminf(fmaxf(dist, 0.f), FRAD);
    env[e] = 0.5f * (cosf(3.14159265358979f * dcl * (1.0f / FRAD)) + 1.f);
  }
}

__global__ void k_concat_z(const float* x, const float* ebuf,
                           const int* src, const int* dst, float* z) {
  int e = blockIdx.x, c = threadIdx.x;
  z[(size_t)e * C3 + c]       = x[(size_t)dst[e] * DD + c];
  z[(size_t)e * C3 + 128 + c] = x[(size_t)src[e] * DD + c];
  z[(size_t)e * C3 + 256 + c] = ebuf[(size_t)e * DD + c];
}

__global__ void k_gate_env(const float* epre, const float* env, float* ge) {
  int e = blockIdx.x, c = threadIdx.x;
  ge[(size_t)e * DD + c] = sigmf(epre[(size_t)e * DD + c]) * env[e];
}

// atomicAdd scatter of [rows,128] into [*,128] via index
__global__ void k_scatter_add(const float* srcb, const int* idx, float* dstb) {
  int n = blockIdx.x, c = threadIdx.x;
  atomicAdd(&dstb[(size_t)idx[n] * DD + c], srcb[(size_t)n * DD + c]);
}

__global__ void k_cart_post_x(const float* aggr, const float* x,
                              const float* g, const float* b, float* hcart) {
  int n = blockIdx.x, c = threadIdx.x;
  hcart[(size_t)n * DD + c] =
      siluf(g[c] * aggr[(size_t)n * DD + c] + b[c]) + x[(size_t)n * DD + c];
}

__global__ void k_cart_post_e(const float* epre, const float* g, const float* b,
                              float* ebuf) {
  int e = blockIdx.x, c = threadIdx.x;
  ebuf[(size_t)e * DD + c] =
      siluf(g[c] * epre[(size_t)e * DD + c] + b[c]) + ebuf[(size_t)e * DD + c];
}

__global__ void k_gn_center(float* Xio, const float* sum, const float* counts,
                            const int* batch, const float* ms, float* var) {
  int n = blockIdx.x, c = threadIdx.x, b = batch[n];
  float mean = sum[(size_t)b * DD + c] * frcp(counts[b]);
  float o = Xio[(size_t)n * DD + c] - mean * ms[c];
  Xio[(size_t)n * DD + c] = o;
  atomicAdd(&var[(size_t)b * DD + c], o * o);
}

__global__ void k_gn_final_add(float* x, const float* Xo, const float* var,
                               const float* counts, const int* batch,
                               const float* w, const float* bb) {
  int n = blockIdx.x, c = threadIdx.x, b = batch[n];
  float v = var[(size_t)b * DD + c] * frcp(counts[b]);
  x[(size_t)n * DD + c] += w[c] * Xo[(size_t)n * DD + c] * rsqrtf(v + FEPS) + bb[c];
}

// build matformer per-(edge,head) rows: m_in = [v_dst|v_src|ea] and
// sig = sigmoid(LN(alpha)) with alpha = tile(q_dst,3)*[k_dst|k_src|ea]/sqrt(3C)
__global__ void k_mat_build(const float* qb, const float* kb, const float* vb,
                            const float* eab, const int* src, const int* dst,
                            int e0, const float* lag, const float* lab,
                            float* m_in, float* sig) {
  __shared__ float red[128];
  int row = blockIdx.x;                 // chunk-local (edge,head) row
  int el = row >> 2, h = row & 3;
  int e = e0 + el;
  int d = dst[e], s = src[e];
  const float* qd = qb + (size_t)d * HC + h * CC;
  const float* kd = kb + (size_t)d * HC + h * CC;
  const float* ks = kb + (size_t)s * HC + h * CC;
  const float* vd = vb + (size_t)d * HC + h * CC;
  const float* vs = vb + (size_t)s * HC + h * CC;
  const float* ee = eab + (size_t)el * HC + h * CC;
  int c = threadIdx.x;
  const float scale = 0.05103103630798288f;   // 1/sqrt(384)
  float q0 = qd[c];
  float a0 = q0 * kd[c] * scale;
  float a1 = q0 * ks[c] * scale;
  float a2 = q0 * ee[c] * scale;
  float* mo = m_in + (size_t)row * C3;
  mo[c] = vd[c]; mo[128 + c] = vs[c]; mo[256 + c] = ee[c];
  const float inv3C = 1.0f / (float)C3;
  float mean = bredsum(a0 + a1 + a2, red) * inv3C;
  float d0 = a0 - mean, d1 = a1 - mean, d2 = a2 - mean;
  float rstd = rsqrtf(bredsum(d0 * d0 + d1 * d1 + d2 * d2, red) * inv3C + FEPS);
  float* so = sig + (size_t)row * C3;
  so[c]       = sigmf(lag[c]       * d0 * rstd + lab[c]);
  so[128 + c] = sigmf(lag[128 + c] * d1 * rstd + lab[128 + c]);
  so[256 + c] = sigmf(lag[256 + c] * d2 * rstd + lab[256 + c]);
}

// LN over 128 then scatter-add (1/H) into out accumulator at dst
__global__ void k_lnm_scatter(const float* m2, const int* dst, int e0,
                              const float* g, const float* b, float* outacc) {
  __shared__ float red[128];
  int row = blockIdx.x;
  int e = e0 + (row >> 2);
  const float* x = m2 + (size_t)row * DD;
  int c = threadIdx.x;
  float v = x[c];
  const float invD = 1.0f / (float)DD;
  float mean = bredsum(v, red) * invD;
  float dv = v - mean;
  float rstd = rsqrtf(bredsum(dv * dv, red) * invD + FEPS);
  atomicAdd(&outacc[(size_t)dst[e] * DD + c], 0.25f * (g[c] * dv * rstd + b[c]));
}

__global__ void k_mat_post(const float* outacc, const float* g, const float* b,
                           float* outs) {
  int n = blockIdx.x, c = threadIdx.x;
  outs[(size_t)n * DD + c] = siluf(g[c] * outacc[(size_t)n * DD + c] + b[c]);
}

__global__ void k_beta(const float* outs, const float* xr, const float* bw,
                       float* hmat) {
  __shared__ float red[128];
  int n = blockIdx.x, c = threadIdx.x;
  float o = outs[(size_t)n * DD + c];
  float r = xr[(size_t)n * DD + c];
  float dot = bredsum(o * bw[c] + r * bw[128 + c] + (o - r) * bw[256 + c], red);
  float beta = sigmf(dot);
  hmat[(size_t)n * DD + c] = beta * r + (1.f - beta) * o;
}

// ---------------- Set2Set ----------------
__global__ void k_s2s_cell(const float* qstar, const float* wih, const float* whh,
                           const float* bih, const float* bhh, float* hh, float* cc) {
  int g0 = blockIdx.x, d = threadIdx.x;
  float gi = bih[d] + bhh[d];
  float gf = bih[128 + d] + bhh[128 + d];
  float gg = bih[256 + d] + bhh[256 + d];
  float go = bih[384 + d] + bhh[384 + d];
  for (int k = 0; k < 256; ++k) {
    float qs = qstar[(size_t)g0 * 256 + k];
    gi += qs * wih[(size_t)d * 256 + k];
    gf += qs * wih[(size_t)(128 + d) * 256 + k];
    gg += qs * wih[(size_t)(256 + d) * 256 + k];
    go += qs * wih[(size_t)(384 + d) * 256 + k];
  }
  for (int k = 0; k < 128; ++k) {
    float hv = hh[(size_t)g0 * DD + k];
    gi += hv * whh[(size_t)d * DD + k];
    gf += hv * whh[(size_t)(128 + d) * DD + k];
    gg += hv * whh[(size_t)(256 + d) * DD + k];
    go += hv * whh[(size_t)(384 + d) * DD + k];
  }
  float ccv = sigmf(gf) * cc[(size_t)g0 * DD + d] + sigmf(gi) * tanhf(gg);
  cc[(size_t)g0 * DD + d] = ccv;
  hh[(size_t)g0 * DD + d] = sigmf(go) * tanhf(ccv);
}

__global__ void k_s2s_ener(const float* x, const float* hh, const int* batch,
                           float* ener) {
  __shared__ float red[128];
  int n = blockIdx.x, c = threadIdx.x, b = batch[n];
  float s = bredsum(x[(size_t)n * DD + c] * hh[(size_t)b * DD + c], red);
  if (c == 0) ener[n] = s;
}

__device__ __forceinline__ void atomicMaxF(float* a, float v) {
  int* ai = (int*)a;
  int old = *ai;
  while (__int_as_float(old) < v) {
    int assumed = old;
    old = atomicCAS(ai, assumed, __float_as_int(v));
    if (old == assumed) break;
  }
}

__global__ void k_seg_max(const float* ener, const int* batch, float* emax) {
  int i = blockIdx.x * 256 + threadIdx.x;
  if (i < NN) atomicMaxF(&emax[batch[i]], ener[i]);
}

__global__ void k_s2s_a(const float* ener, const int* batch, const float* emax,
                        float* a, float* asum) {
  int i = blockIdx.x * 256 + threadIdx.x;
  if (i < NN) {
    float v = __expf(ener[i] - emax[batch[i]]);
    a[i] = v;
    atomicAdd(&asum[batch[i]], v);
  }
}

__global__ void k_s2s_r(const float* x, const float* a, const float* asum,
                        const int* batch, float* r) {
  int n = blockIdx.x, c = threadIdx.x, b = batch[n];
  atomicAdd(&r[(size_t)b * DD + c], a[n] * frcp(asum[b]) * x[(size_t)n * DD + c]);
}

__global__ void k_s2s_qstar(const float* hh, const float* r, float* qstar) {
  int g0 = blockIdx.x, c = threadIdx.x;
  qstar[(size_t)g0 * 256 + c]       = hh[(size_t)g0 * DD + c];
  qstar[(size_t)g0 * 256 + 128 + c] = r[(size_t)g0 * DD + c];
}

__global__ void k_fc(const float* qstar, const float* W1, const float* b1,
                     const float* W2, const float* b2, float* out) {
  __shared__ float red[128];
  int g0 = blockIdx.x, c = threadIdx.x;
  float acc = b1[c];
  for (int k = 0; k < 256; ++k)
    acc += qstar[(size_t)g0 * 256 + k] * W1[(size_t)k * DD + c];
  float t = siluf(acc);
  float s = bredsum(t * W2[c], red);
  if (c == 0) out[g0] = s + b2[0];
}

// ---------------------------------------------------------------------------
// Host orchestration
// ---------------------------------------------------------------------------
struct LayerP {
  const float *g1w, *g1b, *g2w, *g2b, *a1w, *a1b, *a2w, *a2b, *bnxg, *bnxb, *bneg, *bneb;
  const float *qw, *qb, *kw, *kb, *vw, *vb, *ew, *eb, *muw, *mub, *mlw, *mlb;
  const float *lag, *lab, *lmg, *lmb, *bng, *bnb, *skw, *skb, *bw;
  const float *gn1ms, *gn1w, *gn1b, *gn2ms, *gn2w, *gn2b;
};

static inline void gemm(hipStream_t s, const float* A, int lda, const float* W,
                        const float* bias, float* Y, const float* mul,
                        int M, int K, int Nc, int act) {
  dim3 g(Nc / 64, M / 64);
  dim3 b(128);
  if (mul)            k_gemm<0, true ><<<g, b, 0, s>>>(A, lda, W, bias, Y, mul, M, K, Nc);
  else if (act == 1)  k_gemm<1, false><<<g, b, 0, s>>>(A, lda, W, bias, Y, nullptr, M, K, Nc);
  else if (act == 2)  k_gemm<2, false><<<g, b, 0, s>>>(A, lda, W, bias, Y, nullptr, M, K, Nc);
  else                k_gemm<0, false><<<g, b, 0, s>>>(A, lda, W, bias, Y, nullptr, M, K, Nc);
}

extern "C" void kernel_launch(void* const* d_in, const int* in_sizes, int n_in,
                              void* d_out, int out_size, void* d_ws, size_t ws_size,
                              hipStream_t stream) {
  (void)in_sizes; (void)n_in; (void)out_size; (void)ws_size;

  // ---- parse inputs in setup_inputs() insertion order ----
  int ii = 0;
  auto F = [&]() { return (const float*)d_in[ii++]; };
  const float* emb   = F();
  const float* megw  = F(); const float* megb = F();
  const float* gatew = F(); const float* gateb = F();
  const float* o1w   = F(); const float* o1b  = F();
  const float* lng   = F(); const float* lnb  = F();
  const float* o2w   = F(); const float* o2b  = F();
  const float* r1w   = F(); const float* r1b  = F();
  const float* r2w   = F(); const float* r2b  = F();
  const float* wih   = F(); const float* whh  = F();
  const float* bih   = F(); const float* bhh  = F();
  const float* f1w   = F(); const float* f1b  = F();
  const float* f2w   = F(); const float* f2b  = F();
  LayerP L[NLAY];
  for (int l = 0; l < NLAY; ++l) {
    LayerP& p = L[l];
    p.g1w = F(); p.g1b = F(); p.g2w = F(); p.g2b = F();
    p.a1w = F(); p.a1b = F(); p.a2w = F(); p.a2b = F();
    p.bnxg = F(); p.bnxb = F(); p.bneg = F(); p.bneb = F();
    p.qw = F(); p.qb = F(); p.kw = F(); p.kb = F();
    p.vw = F(); p.vb = F(); p.ew = F(); p.eb = F();
    p.muw = F(); p.mub = F(); p.mlw = F(); p.mlb = F();
    p.lag = F(); p.lab = F(); p.lmg = F(); p.lmb = F();
    p.bng = F(); p.bnb = F(); p.skw = F(); p.skb = F();
    p.bw = F();
    p.gn1ms = F(); p.gn1w = F(); p.gn1b = F();
    p.gn2ms = F(); p.gn2w = F(); p.gn2b = F();
  }
  const int*   an    = (const int*)d_in[ii++];
  const float* mge   = F();
  const float* evec  = F();
  const int*   eidx  = (const int*)d_in[ii++];
  const int*   batch = (const int*)d_in[ii++];
  const int* src = eidx;
  const int* dst = eidx + NE;

  // ---- workspace layout (f32 bump allocator) ----
  float* wsf = (float*)d_ws;
  size_t off = 0;
  auto alloc = [&](size_t n) { float* p = wsf + off; off += n; return p; };
  float* counts = alloc(NGR);
  float* xbuf   = alloc((size_t)NN * DD);
  float* cat    = alloc((size_t)NN * 256);
  float* gbuf   = alloc((size_t)NN * DD);
  float* fused  = alloc((size_t)NN * DD);
  float* t1     = alloc((size_t)NN * DD);
  float* t2     = alloc((size_t)NN * DD);
  float* ebuf   = alloc((size_t)NE * DD);
  float* env    = alloc((size_t)NE);
  float* rbff   = alloc((size_t)NE * DD);
  float* zbuf   = alloc((size_t)NE * C3);
  float* tEa    = alloc((size_t)NE * DD);   // g1/a1 silu out
  float* tEb    = alloc((size_t)NE * DD);   // e_pre
  float* tEc    = alloc((size_t)NE * DD);   // gate*env
  float* tEd    = alloc((size_t)NE * DD);   // msg
  float* aggr   = alloc((size_t)NN * DD);
  float* hcart  = alloc((size_t)NN * DD);
  float* gnsum  = alloc((size_t)NGR * DD);
  float* gnvar  = alloc((size_t)NGR * DD);
  float* qn     = alloc((size_t)NN * HC);
  float* kn     = alloc((size_t)NN * HC);
  float* vn     = alloc((size_t)NN * HC);
  float* eab    = alloc((size_t)CE * HC);
  float* m_in   = alloc((size_t)CE * NH * C3);
  float* sigb   = alloc((size_t)CE * NH * C3);
  float* m1     = alloc((size_t)CE * NH * C3);
  float* m2     = alloc((size_t)CE * NH * DD);
  float* outacc = alloc((size_t)NN * DD);
  float* outs   = alloc((size_t)NN * DD);
  float* xr     = alloc((size_t)NN * DD);
  float* hmat   = alloc((size_t)NN * DD);
  float* qstar  = alloc((size_t)NGR * 256);
  float* hhb    = alloc((size_t)NGR * DD);
  float* ccb    = alloc((size_t)NGR * DD);
  float* ener   = alloc((size_t)NN);
  float* abuf   = alloc((size_t)NN);
  float* emax   = alloc((size_t)NGR);
  float* asum   = alloc((size_t)NGR);
  float* rbuf   = alloc((size_t)NGR * DD);

  auto fill = [&](float* p, float v, long long n) {
    int g = (int)((n + 255) / 256); if (g > 4096) g = 4096;
    k_fill<<<g, 256, 0, stream>>>(p, v, n);
  };

  // ---- counts ----
  fill(counts, 0.f, NGR);
  k_counts<<<(NN + 255) / 256, 256, 0, stream>>>(batch, counts);
  k_counts_clamp<<<1, 64, 0, stream>>>(counts);

  // ---- atom encoder ----
  k_atom_pre<<<NN, 128, 0, stream>>>(an, mge, emb, megw, megb, cat);
  gemm(stream, cat, 256, gatew, gateb, gbuf, nullptr, NN, 256, DD, 2);   // sigmoid
  k_atom_fuse<<<NN, 128, 0, stream>>>(gbuf, cat, fused);
  gemm(stream, fused, DD, o1w, o1b, t1, nullptr, NN, DD, DD, 1);         // silu
  k_ln_row<<<NN, 128, 0, stream>>>(t1, t2, lng, lnb, DD, 0);
  gemm(stream, t2, DD, o2w, o2b, xbuf, nullptr, NN, DD, DD, 0);

  // ---- RBF edge embedding ----
  k_rbf<<<NE, 128, 0, stream>>>(evec, rbff, env);
  gemm(stream, rbff, DD, r1w, r1b, tEa, nullptr, NE, DD, DD, 1);
  gemm(stream, tEa, DD, r2w, r2b, ebuf, nullptr, NE, DD, DD, 0);

  auto graph_norm_add = [&](float* hbuf, const float* ms, const float* w,
                            const float* b) {
    fill(gnsum, 0.f, (long long)NGR * DD);
    fill(gnvar, 0.f, (long long)NGR * DD);
    k_scatter_add<<<NN, 128, 0, stream>>>(hbuf, batch, gnsum);
    k_gn_center<<<NN, 128, 0, stream>>>(hbuf, gnsum, counts, batch, ms, gnvar);
    k_gn_final_add<<<NN, 128, 0, stream>>>(xbuf, hbuf, gnvar, counts, batch, w, b);
  };

  // ---- layers ----
  for (int l = 0; l < NLAY; ++l) {
    const LayerP& p = L[l];
    // CartNet
    k_concat_z<<<NE, 128, 0, stream>>>(xbuf, ebuf, src, dst, zbuf);
    gemm(stream, zbuf, C3, p.g1w, p.g1b, tEa, nullptr, NE, C3, DD, 1);
    gemm(stream, tEa, DD, p.g2w, p.g2b, tEb, nullptr, NE, DD, DD, 0);   // e_pre
    gemm(stream, zbuf, C3, p.a1w, p.a1b, tEa, nullptr, NE, C3, DD, 1);
    k_gate_env<<<NE, 128, 0, stream>>>(tEb, env, tEc);
    gemm(stream, tEa, DD, p.a2w, p.a2b, tEd, tEc, NE, DD, DD, 0);       // msg
    fill(aggr, 0.f, (long long)NN * DD);
    k_scatter_add<<<NE, 128, 0, stream>>>(tEd, dst, aggr);
    k_cart_post_x<<<NN, 128, 0, stream>>>(aggr, xbuf, p.bnxg, p.bnxb, hcart);
    k_cart_post_e<<<NE, 128, 0, stream>>>(tEb, p.bneg, p.bneb, ebuf);
    graph_norm_add(hcart, p.gn1ms, p.gn1w, p.gn1b);

    // Matformer
    gemm(stream, xbuf, DD, p.qw, p.qb, qn, nullptr, NN, DD, HC, 0);
    gemm(stream, xbuf, DD, p.kw, p.kb, kn, nullptr, NN, DD, HC, 0);
    gemm(stream, xbuf, DD, p.vw, p.vb, vn, nullptr, NN, DD, HC, 0);
    fill(outacc, 0.f, (long long)NN * DD);
    for (int ch = 0; ch < NCHUNK; ++ch) {
      int e0 = ch * CE;
      gemm(stream, ebuf + (size_t)e0 * DD, DD, p.ew, p.eb, eab, nullptr,
           CE, DD, HC, 0);
      k_mat_build<<<CE * NH, 128, 0, stream>>>(qn, kn, vn, eab, src, dst, e0,
                                               p.lag, p.lab, m_in, sigb);
      gemm(stream, m_in, C3, p.muw, p.mub, m1, sigb, CE * NH, C3, C3, 0);
      gemm(stream, m1, C3, p.mlw, p.mlb, m2, nullptr, CE * NH, C3, DD, 0);
      k_lnm_scatter<<<CE * NH, 128, 0, stream>>>(m2, dst, e0, p.lmg, p.lmb,
                                                 outacc);
    }
    k_mat_post<<<NN, 128, 0, stream>>>(outacc, p.bng, p.bnb, outs);
    gemm(stream, xbuf, DD, p.skw, p.skb, xr, nullptr, NN, DD, DD, 0);
    k_beta<<<NN, 128, 0, stream>>>(outs, xr, p.bw, hmat);
    graph_norm_add(hmat, p.gn2ms, p.gn2w, p.gn2b);
  }

  // ---- Set2Set readout ----
  fill(qstar, 0.f, (long long)NGR * 256);
  fill(hhb, 0.f, (long long)NGR * DD);
  fill(ccb, 0.f, (long long)NGR * DD);
  for (int s = 0; s < NSTEP; ++s) {
    k_s2s_cell<<<NGR, 128, 0, stream>>>(qstar, wih, whh, bih, bhh, hhb, ccb);
    k_s2s_ener<<<NN, 128, 0, stream>>>(xbuf, hhb, batch, ener);
    fill(emax, -3.0e38f, NGR);
    k_seg_max<<<(NN + 255) / 256, 256, 0, stream>>>(ener, batch, emax);
    fill(asum, 0.f, NGR);
    k_s2s_a<<<(NN + 255) / 256, 256, 0, stream>>>(ener, batch, emax, abuf, asum);
    fill(rbuf, 0.f, (long long)NGR * DD);
    k_s2s_r<<<NN, 128, 0, stream>>>(xbuf, abuf, asum, batch, rbuf);
    k_s2s_qstar<<<NGR, 128, 0, stream>>>(hhb, rbuf, qstar);
  }

  // ---- final MLP ----
  k_fc<<<NGR, 128, 0, stream>>>(qstar, f1w, f1b, f2w, f2b, (float*)d_out);
}
